// NunchakuGEGLU_3186865734260
// MI455X (gfx1250) — 8-bit matrix operand order repaired and checked by reading, compiled, not run on hardware
//
#include <hip/hip_runtime.h>
#include <math.h>
#include <stdint.h>

typedef int   v8i __attribute__((ext_vector_type(8)));
typedef float v8f __attribute__((ext_vector_type(8)));
typedef float v2f __attribute__((ext_vector_type(2)));

#define DIM_IN   3072
#define DIM_INW  (DIM_IN/4)
#define DIM_OUT2 24576
#define DIM_OUT  12288
#define SEQ      4096
#define GROUP    64
#define GROUPS   48
#define RANK     32
#define NTILES   (DIM_OUT/16)

#define MBLK     64
#define MT       4
#define LDS_STRIDE 18
#define LDS_BUF  (MBLK * LDS_STRIDE)

#define QX_OFF   0ull
#define QW_OFF   (QX_OFF + (size_t)SEQ*DIM_IN)
#define SX_OFF   (QW_OFF + (size_t)DIM_OUT2*DIM_IN)
#define SW_OFF   (SX_OFF + (size_t)SEQ*4)
#define T_OFF    (SW_OFF + (size_t)DIM_OUT2*GROUPS*4)

__device__ __forceinline__ int quant1(float v, float inv) {
    int q = (int)rintf(v * inv);
    q = q < -8 ? -8 : (q > 7 ? 7 : q);
    return q & 0xFF;
}
__device__ __forceinline__ int pack4(float a, float b, float c, float d, float inv) {
    return quant1(a, inv) | (quant1(b, inv) << 8) |
           (quant1(c, inv) << 16) | (quant1(d, inv) << 24);
}

__device__ __forceinline__ void async_g2l_b32(unsigned lds_off, const int* gptr) {
    asm volatile("global_load_async_to_lds_b32 %0, %1, off"
                 :: "v"(lds_off), "v"(gptr) : "memory");
}
__device__ __forceinline__ void wait_async0() {
    asm volatile("s_wait_asynccnt 0x0" ::: "memory");
}

__global__ void qx_kernel(const float* __restrict__ x, char* __restrict__ qx,
                          float* __restrict__ sx) {
    const int m = blockIdx.x, t = threadIdx.x;
    const float* row = x + (size_t)m * DIM_IN;
    float mx = 0.f;
#pragma unroll
    for (int i = 0; i < 12; ++i) mx = fmaxf(mx, fabsf(row[t * 12 + i]));
    __shared__ float red[256];
    red[t] = mx;
    __syncthreads();
    for (int off = 128; off > 0; off >>= 1) {
        if (t < off) red[t] = fmaxf(red[t], red[t + off]);
        __syncthreads();
    }
    const float s = fmaxf(red[0] * (1.0f / 7.0f), 1e-8f);
    if (t == 0) sx[m] = s;
    const float inv = 1.0f / s;
    int* q = (int*)(qx + (size_t)m * DIM_IN + t * 12);
#pragma unroll
    for (int p = 0; p < 3; ++p) {
        const float* b = row + t * 12 + p * 4;
        q[p] = pack4(b[0], b[1], b[2], b[3], inv);
    }
}

__global__ void qw_kernel(const float* __restrict__ w, char* __restrict__ qw,
                          float* __restrict__ sw) {
    const size_t g = (size_t)blockIdx.x * blockDim.x + threadIdx.x;
    const float4* base = (const float4*)(w + g * GROUP);
    float mx = 0.f;
#pragma unroll
    for (int i = 0; i < 16; ++i) {
        float4 v = base[i];
        mx = fmaxf(mx, fmaxf(fmaxf(fabsf(v.x), fabsf(v.y)),
                             fmaxf(fabsf(v.z), fabsf(v.w))));
    }
    const float s = fmaxf(mx * (1.0f / 7.0f), 1e-8f);
    sw[g] = s;
    const float inv = 1.0f / s;
    int* q = (int*)(qw + g * GROUP);
#pragma unroll
    for (int i = 0; i < 16; ++i) {
        float4 v = base[i];
        q[i] = pack4(v.x, v.y, v.z, v.w, inv);
    }
}

__global__ void lorat_kernel(const float* __restrict__ x, const float* __restrict__ ld,
                             float* __restrict__ t) {
    const int m = blockIdx.x, r = threadIdx.x;
    const float* xr = x + (size_t)m * DIM_IN;
    const float* lr = ld + (size_t)r * DIM_IN;
    float acc = 0.f;
    for (int k = 0; k < DIM_IN; k += 4) {
        float4 xv = *(const float4*)(xr + k);
        float4 lv = *(const float4*)(lr + k);
        acc += xv.x * lv.x + xv.y * lv.y + xv.z * lv.z + xv.w * lv.w;
    }
    t[(size_t)m * RANK + r] = acc;
}

__global__ void __launch_bounds__(256)
main_kernel(const char* __restrict__ qx, const float* __restrict__ sx,
            const char* __restrict__ qw, const float* __restrict__ sw,
            const float* __restrict__ t, const float* __restrict__ up,
            float* __restrict__ out) {
    __shared__ int ldsA[2 * LDS_BUF];

    const int tid  = threadIdx.x;
    const int lane = tid & 31;
    const int wv   = tid >> 5;
    const int half = lane >> 4;
    const int l16  = lane & 15;

    const int mblk  = blockIdx.x / (NTILES / 8);
    const int nblk  = blockIdx.x % (NTILES / 8);
    const int mbase = mblk * MBLK;
    const int ntile = nblk * 8 + wv;
    const int n_h = ntile * 16 + l16;
    const int n_g = n_h + DIM_OUT;

    const char* bhr = qw + (size_t)n_h * DIM_IN;
    const char* bgr = qw + (size_t)n_g * DIM_IN;
    const int*  qxi = (const int*)qx;

    const unsigned ldsBase = (unsigned)(uintptr_t)(&ldsA[0]);
    const int srow = tid >> 4;
    const int scol = tid & 15;
    const int* gsrc0 = qxi + (size_t)(mbase + srow) * DIM_INW + scol;

    v8f fh[MT], fg[MT];
#pragma unroll
    for (int mt = 0; mt < MT; ++mt) { fh[mt] = (v8f){}; fg[mt] = (v8f){}; }

#pragma unroll
    for (int i = 0; i < 4; ++i) {
        const unsigned loff = ldsBase +
            (unsigned)(((srow + i * 16) * LDS_STRIDE + scol) * 4);
        async_g2l_b32(loff, gsrc0 + (size_t)(i * 16) * DIM_INW);
    }
    wait_async0();
    __syncthreads();

    for (int g = 0; g < GROUPS; ++g) {
        const int cur = g & 1;
        if (g + 1 < GROUPS) {
            const int nxt = cur ^ 1;
#pragma unroll
            for (int i = 0; i < 4; ++i) {
                const unsigned loff = ldsBase + (unsigned)((nxt * LDS_BUF +
                    (srow + i * 16) * LDS_STRIDE + scol) * 4);
                async_g2l_b32(loff, gsrc0 + (size_t)(i * 16) * DIM_INW + (g + 1) * 16);
            }
            __builtin_prefetch(bhr + (g + 1) * GROUP, 0, 3);
            __builtin_prefetch(bgr + (g + 1) * GROUP, 0, 3);
        }
        const int k0 = g * GROUP;
        v8i Bh, Bg;
#pragma unroll
        for (int v = 0; v < 8; ++v) {
            const int kb = k0 + (v >> 1) * 16 + half * 8 + (v & 1) * 4;
            Bh[v] = *(const int*)(bhr + kb);
            Bg[v] = *(const int*)(bgr + kb);
        }
        const float swh = sw[(size_t)n_h * GROUPS + g];
        const float swg = sw[(size_t)n_g * GROUPS + g];
        const int abase = cur * LDS_BUF;
        v8i afr[MT];
#pragma unroll
        for (int mt = 0; mt < MT; ++mt)
#pragma unroll
            for (int v = 0; v < 8; ++v)
                afr[mt][v] = ldsA[abase + (mt * 16 + l16) * LDS_STRIDE +
                                  ((v >> 1) * 4 + (v & 1) + half * 2)];
#pragma unroll
        for (int mt = 0; mt < MT; ++mt) {
            v8i c0 = {};
            v8i ch = __builtin_amdgcn_wmma_i32_16x16x64_iu8(true, afr[mt], true, Bh, c0, false, false);
            v8i cg = __builtin_amdgcn_wmma_i32_16x16x64_iu8(true, afr[mt], true, Bg, c0, false, false);
#pragma unroll
            for (int v = 0; v < 8; ++v) {
                fh[mt][v] += (float)ch[v] * swh;
                fg[mt][v] += (float)cg[v] * swg;
            }
        }
        if (g + 1 < GROUPS) {
            wait_async0();
            __syncthreads();
        }
    }

    v2f Buh[8], Bug[8];
#pragma unroll
    for (int rc = 0; rc < 8; ++rc) {
        Buh[rc] = *(const v2f*)(up + (size_t)n_h * RANK + rc * 4 + 2 * half);
        Bug[rc] = *(const v2f*)(up + (size_t)n_g * RANK + rc * 4 + 2 * half);
    }

#pragma unroll
    for (int mt = 0; mt < MT; ++mt) {
        const int mrow = mbase + mt * 16;
        v8f ch = fh[mt], cg = fg[mt];
#pragma unroll
        for (int v = 0; v < 8; ++v) {
            const float sv = sx[mrow + v + 8 * half];
            ch[v] *= sv;
            cg[v] *= sv;
        }
        const float* trow = t + (size_t)(mrow + l16) * RANK;
#pragma unroll
        for (int rc = 0; rc < 8; ++rc) {
            v2f a = *(const v2f*)(trow + rc * 4 + 2 * half);
            ch = __builtin_amdgcn_wmma_f32_16x16x4_f32(false, a, false, Buh[rc],
                                                       (short)0, ch, false, false);
            cg = __builtin_amdgcn_wmma_f32_16x16x4_f32(false, a, false, Bug[rc],
                                                       (short)0, cg, false, false);
        }
#pragma unroll
        for (int v = 0; v < 8; ++v) {
            const int m = mrow + v + 8 * half;
            const float gt = cg[v];
            const float gl = 0.5f * gt * (1.0f + erff(gt * 0.70710678118654752f));
            out[(size_t)m * DIM_OUT + n_h] = ch[v] * gl;
        }
    }
}

extern "C" void kernel_launch(void* const* d_in, const int* in_sizes, int n_in,
                              void* d_out, int out_size, void* d_ws, size_t ws_size,
                              hipStream_t stream) {
    const float* x  = (const float*)d_in[0];
    const float* w  = (const float*)d_in[1];
    const float* ld = (const float*)d_in[2];
    const float* up = (const float*)d_in[3];
    float* out = (float*)d_out;
    char*  ws  = (char*)d_ws;

    char*  qx = ws + QX_OFF;
    char*  qw = ws + QW_OFF;
    float* sx = (float*)(ws + SX_OFF);
    float* sw = (float*)(ws + SW_OFF);
    float* t  = (float*)(ws + T_OFF);

    qx_kernel<<<SEQ, 256, 0, stream>>>(x, qx, sx);
    qw_kernel<<<(DIM_OUT2 * GROUPS) / 256, 256, 0, stream>>>(w, qw, sw);
    lorat_kernel<<<SEQ, 32, 0, stream>>>(x, ld, t);
    main_kernel<<<(SEQ / MBLK) * (NTILES / 8), 256, 0, stream>>>(qx, sx, qw, sw, t, up, out);
}
